// GNNClassifier_88648124990108
// MI455X (gfx1250) — compile-verified, matrix operand order checked by reading, not run on hardware
//
#include <hip/hip_runtime.h>
#include <stdint.h>

// ---------------------------------------------------------------------------
// GNN SAGE classifier for MI455X (gfx1250, wave32).
//  - edge scatter-adds via global_atomic_add_f32 (L2-resident: node state fits
//    easily in the 192MB L2, so atomics and gathers never hit HBM)
//  - dense per-node transforms via v_wmma_f32_16x16x32_f16 (f16 in, f32 acc)
//  - A-tile staging via global_load_async_to_lds_b128 (ASYNCcnt path, no VGPR
//    round-trip), mean-scaling folded into the per-lane fragment gather
// ---------------------------------------------------------------------------

typedef __attribute__((ext_vector_type(16))) _Float16 v16h;
typedef __attribute__((ext_vector_type(8)))  float    v8f;
typedef __attribute__((ext_vector_type(8)))  uint32_t v8u;

#define HID 64

// ISA 16-bit A-matrix 16x32 layout (CDNA5 ISA §7.12.2):
// lanes 0-15 : halves 0..7 -> K=0..7,  halves 8..15 -> K=16..23
// lanes 16-31: halves 0..7 -> K=8..15, halves 8..15 -> K=24..31
__device__ __forceinline__ int kmap(int i, int lane) {
    int k = (i < 8) ? i : (i + 8);
    if (lane >= 16) k += 8;
    return k;
}

__device__ __forceinline__ v16h as_v16h(v8u u) {
    union { v8u u; v16h h; } x; x.u = u; return x.h;
}

// Async copy 16B global -> LDS. vdst = LDS byte address (wave-relative; flat
// shared pointers carry it in addr[31:0]), 64-bit global address in VGPR pair.
__device__ __forceinline__ void async_copy_b128(uint32_t lds_addr, const void* gptr) {
    uint64_t ga = (uint64_t)(uintptr_t)gptr;
    asm volatile("global_load_async_to_lds_b128 %0, %1, off"
                 :: "v"(lds_addr), "v"(ga) : "memory");
}

__device__ __forceinline__ void wait_async_zero() {
    asm volatile("s_wait_asynccnt 0x0" ::: "memory");
}

__device__ __forceinline__ uint32_t lds_addr_of(const void* p) {
    return (uint32_t)(uintptr_t)p;   // LDS aperture: addr[31:0] = LDS address
}

// --------------------------- small prep/scatter kernels --------------------

__global__ void embed_gather_k(const int* __restrict__ ids,
                               const float* __restrict__ table,
                               float* __restrict__ x0, int nNodes) {
    int id = blockIdx.x * blockDim.x + threadIdx.x;
    if (id >= nNodes * 32) return;
    int n = id >> 5, e = id & 31;
    x0[id] = table[(size_t)ids[n] * 32 + e];
}

__global__ void degree_k(const int* __restrict__ dst, float* __restrict__ cnt,
                         int nEdges) {
    int e = blockIdx.x * blockDim.x + threadIdx.x;
    if (e < nEdges) atomicAdd(&cnt[dst[e]], 1.0f);
}

// one thread per (edge, feature); E = 1<<shift
__global__ void scatter_add_k(const float* __restrict__ feat,
                              const int* __restrict__ src,
                              const int* __restrict__ dst,
                              float* __restrict__ out, int shift, int total) {
    int id = blockIdx.x * blockDim.x + threadIdx.x;
    if (id >= total) return;
    int e = id >> shift;
    int c = id & ((1 << shift) - 1);
    float v = feat[((size_t)src[e] << shift) + c];
    atomicAdd(&out[((size_t)dst[e] << shift) + c], v);
}

// Pre-pack B fragments (weights) in the per-lane register layout so GEMM waves
// load them with coalesced 128b loads. B[k][n] = Wcat[n][k], Wcat = [Wl | Wr].
// Layout: dword index ((kb*4 + ntile)*32 + lane)*8 + j  packs halves 2j, 2j+1.
__global__ void build_bfrag_k(const float* __restrict__ Wl,
                              const float* __restrict__ Wr,
                              uint32_t* __restrict__ bfrag, int E) {
    int lane = threadIdx.x;          // 32 threads
    int ksteps = (2 * E) / 32;
    for (int kb = 0; kb < ksteps; ++kb)
        for (int t = 0; t < 4; ++t)
            for (int j = 0; j < 8; ++j) {
                int n  = t * 16 + (lane & 15);
                int k0 = kb * 32 + kmap(2 * j + 0, lane);
                int k1 = kb * 32 + kmap(2 * j + 1, lane);
                float f0 = (k0 < E) ? Wl[n * E + k0] : Wr[n * E + (k0 - E)];
                float f1 = (k1 < E) ? Wl[n * E + k1] : Wr[n * E + (k1 - E)];
                union { _Float16 h[2]; uint32_t u; } p;
                p.h[0] = (_Float16)f0;
                p.h[1] = (_Float16)f1;
                bfrag[((size_t)((kb * 4 + t) * 32 + lane)) * 8 + j] = p.u;
            }
}

// --------------------------- fused SAGE GEMM (WMMA) ------------------------
// out[n,:] = relu( [msg[n]/max(cnt,1) | self[n]] @ Wcat^T + bias )
// Block: 128 threads = 4 waves; each wave owns a 16-node tile (64 nodes/blk).
// A tile staged raw into LDS with async copies; the mean scale (and the OOB
// zero-mask) is applied during fragment gather: uniform per K-step.
template <int E>
__global__ void sage_gemm_k(const float* __restrict__ msg,
                            const float* __restrict__ self,
                            const float* __restrict__ cnt,
                            const uint32_t* __restrict__ bfrag,
                            const float* __restrict__ bias,
                            float* __restrict__ out, int nNodes) {
    constexpr int K = 2 * E;
    constexpr int KSTEPS = K / 32;
    constexpr int CHUNKS = 64 * K / 4;      // 16B chunks in the A tile
    __shared__ float At[64 * K];

    const int tid = threadIdx.x;
    const int nodeBase = blockIdx.x * 64;

    // ---- async-stage A tile [64 x K] = [msg | self] (raw, unscaled) ----
    for (int c = tid; c < CHUNKS; c += 128) {   // uniform trip count (no div.)
        int d   = c * 4;                        // dword index in tile
        int row = d / K;
        int k   = d % K;                        // 16B chunk stays in one half
        int node = nodeBase + row;
        if (node > nNodes - 1) node = nNodes - 1;   // clamp; masked at gather
        const float* src = (k < E) ? (msg  + (size_t)node * E + k)
                                   : (self + (size_t)node * E + (k - E));
        async_copy_b128(lds_addr_of(&At[d]), src);
    }
    wait_async_zero();
    __syncthreads();

    const int wave = tid >> 5;
    const int lane = tid & 31;
    const int m = lane & 15;
    const int myNode = nodeBase + wave * 16 + m;
    const int rowBase = (wave * 16 + m) * K;

    // per-lane scale factors: mean half gets 1/max(deg,1); self half gets 1;
    // out-of-range tail rows get 0 (replaces zero-fill the async copy skipped)
    float fm = 0.0f, fs = 0.0f;
    if (myNode < nNodes) {
        fm = 1.0f / fmaxf(cnt[myNode], 1.0f);
        fs = 1.0f;
    }

    v8f acc[4] = {v8f{}, v8f{}, v8f{}, v8f{}};
    const v8u* bp = (const v8u*)bfrag;

    for (int kb = 0; kb < KSTEPS; ++kb) {
        float f = (kb < KSTEPS / 2) ? fm : fs;   // uniform per K-step
        v16h a;
#pragma unroll
        for (int i = 0; i < 16; ++i) {
            int k = kb * 32 + kmap(i, lane);
            a[i] = (_Float16)(At[rowBase + k] * f);
        }
#pragma unroll
        for (int t = 0; t < 4; ++t) {
            v16h b = as_v16h(bp[(kb * 4 + t) * 32 + lane]);
            acc[t] = __builtin_amdgcn_wmma_f32_16x16x32_f16(
                false, a, false, b, (short)0, acc[t], false, false);
        }
    }

    // D layout: lanes 0-15 N=lane, M=r; lanes 16-31 N=lane-16, M=r+8.
    const int mAdd = (lane >> 4) * 8;
    const int col0 = lane & 15;
#pragma unroll
    for (int t = 0; t < 4; ++t) {
        int col = t * 16 + col0;
        float bcol = bias[col];
#pragma unroll
        for (int r = 0; r < 8; ++r) {
            int node = nodeBase + wave * 16 + mAdd + r;
            if (node < nNodes)
                out[(size_t)node * HID + col] = fmaxf(acc[t][r] + bcol, 0.0f);
        }
    }
}

// --------------------------- pooling + classifier --------------------------

__global__ void pool_scatter_k(const float* __restrict__ h2,
                               const int* __restrict__ batch,
                               float* __restrict__ gsum,
                               float* __restrict__ gcnt, int nNodes) {
    int id = blockIdx.x * blockDim.x + threadIdx.x;
    if (id >= nNodes * HID) return;
    int n = id >> 6, h = id & 63;
    int b = batch[n];
    atomicAdd(&gsum[b * HID + h], h2[id]);
    if (h == 0) atomicAdd(&gcnt[b], 1.0f);
}

__global__ void classify_k(const float* __restrict__ gsum,
                           const float* __restrict__ gcnt,
                           const float* __restrict__ Wlin,
                           const float* __restrict__ blin,
                           float* __restrict__ out, int nGraphs) {
    int t = blockIdx.x * blockDim.x + threadIdx.x;
    if (t >= nGraphs * 2) return;
    int g = t >> 1, c = t & 1;
    float inv = 1.0f / fmaxf(gcnt[g], 1.0f);
    float acc = 0.0f;
    for (int h = 0; h < HID; ++h)
        acc += gsum[g * HID + h] * inv * Wlin[c * HID + h];
    out[t] = acc + blin[c];
}

// ---------------------------------------------------------------------------

extern "C" void kernel_launch(void* const* d_in, const int* in_sizes, int n_in,
                              void* d_out, int out_size, void* d_ws, size_t ws_size,
                              hipStream_t stream) {
    (void)n_in; (void)ws_size;
    const int*   node_ids = (const int*)d_in[0];
    const int*   edge     = (const int*)d_in[1];
    const int*   batch    = (const int*)d_in[2];
    const float* table    = (const float*)d_in[3];
    const float* W1l      = (const float*)d_in[4];
    const float* W1r      = (const float*)d_in[5];
    const float* b1       = (const float*)d_in[6];
    const float* W2l      = (const float*)d_in[7];
    const float* W2r      = (const float*)d_in[8];
    const float* b2       = (const float*)d_in[9];
    const float* Wlin     = (const float*)d_in[10];
    const float* blin     = (const float*)d_in[11];
    float* out = (float*)d_out;

    const int N    = in_sizes[0];
    const int EDG  = in_sizes[1] / 2;
    const int G    = out_size / 2;
    const int* src = edge;
    const int* dst = edge + EDG;

    // workspace layout (floats)
    float* ws = (float*)d_ws;
    size_t o = 0;
    float* x0   = ws + o; o += (size_t)N * 32;   // embedded features
    float* msgb = ws + o; o += (size_t)N * 64;   // message accumulator (reused)
    float* h1   = ws + o; o += (size_t)N * 64;   // layer-1 output
    float* h2   = ws + o; o += (size_t)N * 64;   // layer-2 output
    float* cnt  = ws + o; o += (size_t)N;        // in-degree
    float* gsum = ws + o; o += (size_t)G * HID;  // graph pooled sums
    float* gcnt = ws + o; o += (size_t)G;        // graph node counts
    o = (o + 7) & ~(size_t)7;                    // 32B align for v8u loads
    uint32_t* bfrag1 = (uint32_t*)(ws + o); o += 2 * 4 * 32 * 8;
    uint32_t* bfrag2 = (uint32_t*)(ws + o); o += 4 * 4 * 32 * 8;

    // zero accumulators (graph-capture-safe async memsets)
    hipMemsetAsync(cnt,  0, (size_t)N * sizeof(float), stream);
    hipMemsetAsync(gsum, 0, (size_t)G * HID * sizeof(float), stream);
    hipMemsetAsync(gcnt, 0, (size_t)G * sizeof(float), stream);

    // pack weight fragments for WMMA B-operand layout
    build_bfrag_k<<<1, 32, 0, stream>>>(W1l, W1r, bfrag1, 32);
    build_bfrag_k<<<1, 32, 0, stream>>>(W2l, W2r, bfrag2, 64);

    // x0 = embed_table[node_ids]
    {
        int total = N * 32;
        embed_gather_k<<<(total + 255) / 256, 256, 0, stream>>>(node_ids, table, x0, N);
    }

    // degree
    degree_k<<<(EDG + 255) / 256, 256, 0, stream>>>(dst, cnt, EDG);

    // ---- layer 1: scatter x0 (E=32), then fused mean+GEMM+bias+relu ----
    hipMemsetAsync(msgb, 0, (size_t)N * 32 * sizeof(float), stream);
    {
        int total = EDG * 32;
        scatter_add_k<<<(total + 255) / 256, 256, 0, stream>>>(x0, src, dst, msgb, 5, total);
    }
    {
        int blocks = (N + 63) / 64;
        sage_gemm_k<32><<<blocks, 128, 0, stream>>>(msgb, x0, cnt, bfrag1, b1, h1, N);
    }

    // ---- layer 2: scatter h1 (E=64), then fused mean+GEMM+bias+relu ----
    hipMemsetAsync(msgb, 0, (size_t)N * 64 * sizeof(float), stream);
    {
        int total = EDG * 64;
        scatter_add_k<<<(total + 255) / 256, 256, 0, stream>>>(h1, src, dst, msgb, 6, total);
    }
    {
        int blocks = (N + 63) / 64;
        sage_gemm_k<64><<<blocks, 128, 0, stream>>>(msgb, h1, cnt, bfrag2, b2, h2, N);
    }

    // ---- global mean pool + classifier ----
    {
        int total = N * HID;
        pool_scatter_k<<<(total + 255) / 256, 256, 0, stream>>>(h2, batch, gsum, gcnt, N);
    }
    classify_k<<<(G * 2 + 255) / 256, 256, 0, stream>>>(gsum, gcnt, Wlin, blin, out, G);
}
